// SCTConv_67405216743976
// MI455X (gfx1250) — matrix operand order repaired and checked by reading, not compiled, not run on hardware
//
#include <hip/hip_runtime.h>
#include <hip/hip_bf16.h>

#define NN 8192
#define HH 128

typedef __bf16 bf16_t;
typedef __attribute__((ext_vector_type(16))) bf16_t v16bf;
typedef __attribute__((ext_vector_type(8)))  bf16_t v8bf;
typedef __attribute__((ext_vector_type(4)))  bf16_t v4bf;
typedef __attribute__((ext_vector_type(8)))  float  v8f;

__device__ __forceinline__ float leaky01(float x) { return x > 0.f ? x : 0.01f * x; }
__device__ __forceinline__ float relu0(float x)   { return x > 0.f ? x : 0.f; }

// ---------------------------------------------------------------------------
// deg[i] = sum_j adj[i][j];  dinv[i] = rsqrt(deg[i] + 1)   (A_hat = adj + I)
// ---------------------------------------------------------------------------
__global__ __launch_bounds__(256) void k_rowsum(const float* __restrict__ adj,
                                                float* __restrict__ deg,
                                                float* __restrict__ dinv) {
  const int row = blockIdx.x;
  __shared__ float red[256];
  float s = 0.f;
  const float* r = adj + (size_t)row * NN;
  for (int j = threadIdx.x; j < NN; j += 256) s += r[j];
  red[threadIdx.x] = s;
  __syncthreads();
  for (int st = 128; st > 0; st >>= 1) {
    if (threadIdx.x < st) red[threadIdx.x] += red[threadIdx.x + st];
    __syncthreads();
  }
  if (threadIdx.x == 0) {
    float d = red[0];
    deg[row]  = d;
    dinv[row] = rsqrtf(d + 1.0f);
  }
}

// ---------------------------------------------------------------------------
// adj fp32 -> bf16 (raw, unscaled; scalings folded elsewhere)
// ---------------------------------------------------------------------------
__global__ __launch_bounds__(256) void k_cvt_adj(const float* __restrict__ adj,
                                                 bf16_t* __restrict__ abf) {
  size_t i = ((size_t)blockIdx.x * 256 + threadIdx.x) * 4;
  float4 v = *(const float4*)(adj + i);
  v4bf o;
  o[0] = (bf16_t)v.x; o[1] = (bf16_t)v.y; o[2] = (bf16_t)v.z; o[3] = (bf16_t)v.w;
  *(v4bf*)(abf + i) = o;
}

// ---------------------------------------------------------------------------
// BT[n][k] = bf16( V[k][n] * scale[k] )   scale = dinv (mode 0) | 1/deg (mode 1)
// grid (HH/32, NN/32), block (32,8). 32x32 LDS tile transpose.
// ---------------------------------------------------------------------------
__global__ __launch_bounds__(256) void k_prep_bt(const float* __restrict__ V,
                                                 const float* __restrict__ dinv,
                                                 const float* __restrict__ deg,
                                                 bf16_t* __restrict__ BT,
                                                 int mode) {
  __shared__ float tile[32][33];
  const int nb = blockIdx.x * 32;
  const int kb = blockIdx.y * 32;
  for (int i = threadIdx.y; i < 32; i += 8) {
    int k = kb + i;
    float s = (mode == 0) ? dinv[k] : (1.0f / deg[k]);
    tile[i][threadIdx.x] = V[(size_t)k * HH + nb + threadIdx.x] * s;
  }
  __syncthreads();
  for (int i = threadIdx.y; i < 32; i += 8) {
    int n = nb + i;
    int k = kb + threadIdx.x;
    BT[(size_t)n * NN + k] = (bf16_t)tile[threadIdx.x][i];
  }
}

// ---------------------------------------------------------------------------
// Out = epilogue( adj_bf16 @ B ), B given transposed+prescaled as BT[HH][NN].
// Block: 128 thr = 4 waves. Wave w computes a 2x2 grid of 16x16 C tiles:
//   rows [blockIdx.x*32, +32), cols [w*32, +32).
// Per K-step (32): 2 A-frags + 2 B-frags -> 4 WMMAs (register-blocked reuse).
// mode 0: out = dr[i]*C + dr[i]^2 * Vsrc[i][n]      (A_norm incl. +I term)
// mode 1: out = 0.5*C + 0.5*Vsrc[i][n]              (lazy walk P)
// ---------------------------------------------------------------------------
__global__ __launch_bounds__(128) void k_gemm(const bf16_t* __restrict__ A,
                                              const bf16_t* __restrict__ BT,
                                              const float* __restrict__ Vsrc,
                                              const float* __restrict__ dinv,
                                              float* __restrict__ Out,
                                              int mode) {
  const int tid   = threadIdx.x;
  const int wave  = tid >> 5;        // col pair 0..3
  const int lane  = tid & 31;
  const int laneM = lane & 15;
  const int half  = lane >> 4;
  const size_t r0 = (size_t)blockIdx.x * 32;

  v8f acc[2][2];
  #pragma unroll
  for (int rt = 0; rt < 2; rt++)
    #pragma unroll
    for (int ct = 0; ct < 2; ct++)
      #pragma unroll
      for (int e = 0; e < 8; e++) acc[rt][ct][e] = 0.f;

  const bf16_t* Arow0 = A + (r0 + laneM) * (size_t)NN;
  const bf16_t* Arow1 = A + (r0 + 16 + laneM) * (size_t)NN;
  const bf16_t* Brow0 = BT + (size_t)(wave * 32 + laneM) * NN;
  const bf16_t* Brow1 = BT + (size_t)(wave * 32 + 16 + laneM) * NN;

  #pragma unroll 2
  for (int kb = 0; kb < NN; kb += 32) {
    // A fragment: lanes 0-15 -> K[kb..kb+7] & K[kb+16..kb+23];
    //             lanes 16-31 -> K[kb+8..kb+15] & K[kb+24..kb+31]
    const int k0 = kb + half * 8;
    v16bf af0, af1;
    {
      v8bf lo = *(const v8bf*)(Arow0 + k0);
      v8bf hi = *(const v8bf*)(Arow0 + k0 + 16);
      #pragma unroll
      for (int e = 0; e < 8; e++) { af0[e] = lo[e]; af0[8 + e] = hi[e]; }
    }
    {
      v8bf lo = *(const v8bf*)(Arow1 + k0);
      v8bf hi = *(const v8bf*)(Arow1 + k0 + 16);
      #pragma unroll
      for (int e = 0; e < 8; e++) { af1[e] = lo[e]; af1[8 + e] = hi[e]; }
    }

    // B fragment: lanes 0-15 -> K[kb..kb+7] & K[kb+16..kb+23]; lanes 16-31 -> K[kb+8..kb+15] & K[kb+24..kb+31]
    const int kB = kb + half * 8;
    v16bf bf0, bf1;
    {
      v8bf lo = *(const v8bf*)(Brow0 + kB);
      v8bf hi = *(const v8bf*)(Brow0 + kB + 16);
      #pragma unroll
      for (int e = 0; e < 8; e++) { bf0[e] = lo[e]; bf0[8 + e] = hi[e]; }
    }
    {
      v8bf lo = *(const v8bf*)(Brow1 + kB);
      v8bf hi = *(const v8bf*)(Brow1 + kB + 16);
      #pragma unroll
      for (int e = 0; e < 8; e++) { bf1[e] = lo[e]; bf1[8 + e] = hi[e]; }
    }

    acc[0][0] = __builtin_amdgcn_wmma_f32_16x16x32_bf16(false, af0, false, bf0,
                                                        (short)0, acc[0][0], false, false);
    acc[0][1] = __builtin_amdgcn_wmma_f32_16x16x32_bf16(false, af0, false, bf1,
                                                        (short)0, acc[0][1], false, false);
    acc[1][0] = __builtin_amdgcn_wmma_f32_16x16x32_bf16(false, af1, false, bf0,
                                                        (short)0, acc[1][0], false, false);
    acc[1][1] = __builtin_amdgcn_wmma_f32_16x16x32_bf16(false, af1, false, bf1,
                                                        (short)0, acc[1][1], false, false);
  }

  // C layout: lanes 0-15: row = base + v, lanes 16-31: row = base + 8 + v; col = laneM
  #pragma unroll
  for (int rt = 0; rt < 2; rt++) {
    #pragma unroll
    for (int ct = 0; ct < 2; ct++) {
      const int col = wave * 32 + ct * 16 + laneM;
      #pragma unroll
      for (int v = 0; v < 8; v++) {
        size_t row = r0 + rt * 16 + half * 8 + v;
        float c = acc[rt][ct][v];
        float o;
        if (mode == 0) {
          float dr = dinv[row];
          o = dr * c + dr * dr * Vsrc[row * HH + col];
        } else {
          o = 0.5f * c + 0.5f * Vsrc[row * HH + col];
        }
        Out[row * HH + col] = o;
      }
    }
  }
}

// ---------------------------------------------------------------------------
// 6-way attention fusion. One block (128 thr) per node.
// ---------------------------------------------------------------------------
__global__ __launch_bounds__(128) void k_fuse(const float* __restrict__ X,
                                              const float* __restrict__ hA,
                                              const float* __restrict__ hA2,
                                              const float* __restrict__ hA3,
                                              const float* __restrict__ px,
                                              const float* __restrict__ p2x,
                                              const float* __restrict__ p4x,
                                              const float* __restrict__ a,
                                              float* __restrict__ hp) {
  const int i = blockIdx.x, k = threadIdx.x;
  const size_t o = (size_t)i * HH + k;
  float x   = X[o];
  float pxv = px[o], p2v = p2x[o], p4v = p4x[o];
  float b[6];
  b[0] = leaky01(hA[o]);
  b[1] = leaky01(hA2[o]);
  b[2] = leaky01(hA3[o]);
  b[3] = fabsf(x - pxv);
  b[4] = fabsf(pxv - p2v);
  b[5] = fabsf(p2v - p4v);
  float a0 = a[k], a1 = a[HH + k];

  __shared__ float red[7][128];
  red[0][k] = relu0(x) * a0;
  #pragma unroll
  for (int s = 0; s < 6; s++) red[s + 1][k] = relu0(b[s]) * a1;
  __syncthreads();
  for (int st = 64; st > 0; st >>= 1) {
    if (k < st) {
      #pragma unroll
      for (int r = 0; r < 7; r++) red[r][k] += red[r][k + st];
    }
    __syncthreads();
  }
  __shared__ float attn[6];
  if (k == 0) {
    float base = red[0][0];
    float e[6], mx = -1e30f;
    #pragma unroll
    for (int s = 0; s < 6; s++) { e[s] = base + red[s + 1][0]; mx = fmaxf(mx, e[s]); }
    float sum = 0.f;
    #pragma unroll
    for (int s = 0; s < 6; s++) { e[s] = expf(e[s] - mx); sum += e[s]; }
    #pragma unroll
    for (int s = 0; s < 6; s++) attn[s] = e[s] / sum;
  }
  __syncthreads();
  float v = 0.f;
  #pragma unroll
  for (int s = 0; s < 6; s++) v += attn[s] * b[s];
  hp[o] = v * (1.0f / 6.0f);
}

// ---------------------------------------------------------------------------
// out[i][o] = leaky( b[o] + sum_k in[i][k] * W[o][k] )   (x @ W.T + b)
// ---------------------------------------------------------------------------
__global__ __launch_bounds__(128) void k_mlp(const float* __restrict__ in,
                                             const float* __restrict__ W,
                                             const float* __restrict__ b,
                                             float* __restrict__ out) {
  const int i = blockIdx.x, o = threadIdx.x;
  __shared__ float row[128];
  row[o] = in[(size_t)i * HH + o];
  __syncthreads();
  float s = b[o];
  const float* w = W + (size_t)o * HH;
  #pragma unroll 8
  for (int k = 0; k < HH; k++) s += row[k] * w[k];
  out[(size_t)i * HH + o] = leaky01(s);
}

// ---------------------------------------------------------------------------
extern "C" void kernel_launch(void* const* d_in, const int* in_sizes, int n_in,
                              void* d_out, int out_size, void* d_ws, size_t ws_size,
                              hipStream_t stream) {
  (void)in_sizes; (void)n_in; (void)out_size; (void)ws_size;
  const float* X   = (const float*)d_in[0];
  const float* adj = (const float*)d_in[1];
  const float* W1  = (const float*)d_in[2];
  const float* b1  = (const float*)d_in[3];
  const float* W2  = (const float*)d_in[4];
  const float* b2  = (const float*)d_in[5];
  const float* a   = (const float*)d_in[6];

  char* w = (char*)d_ws;
  size_t off = 0;
  auto take = [&](size_t bytes) { void* p = w + off; off = (off + bytes + 255) & ~(size_t)255; return p; };

  bf16_t* adj_bf = (bf16_t*)take((size_t)NN * NN * sizeof(bf16_t));   // 128 MB
  bf16_t* BT     = (bf16_t*)take((size_t)HH * NN * sizeof(bf16_t));   // 2 MB
  float*  deg    = (float*)take(NN * sizeof(float));
  float*  dinv   = (float*)take(NN * sizeof(float));
  const size_t NH = (size_t)NN * HH * sizeof(float);                  // 4 MB each
  float* hA  = (float*)take(NH);
  float* hA2 = (float*)take(NH);
  float* hA3 = (float*)take(NH);
  float* px  = (float*)take(NH);
  float* p2x = (float*)take(NH);
  float* p3x = (float*)take(NH);
  float* p4x = (float*)take(NH);
  float* hp  = (float*)take(NH);
  float* tmp = (float*)take(NH);

  // prep
  k_rowsum<<<NN, 256, 0, stream>>>(adj, deg, dinv);
  k_cvt_adj<<<(int)(((size_t)NN * NN) / (256 * 4)), 256, 0, stream>>>(adj, adj_bf);

  dim3 pgrid(HH / 32, NN / 32), pblk(32, 8);
  auto pass = [&](const float* Vsrc, float* Out, int mode) {
    k_prep_bt<<<pgrid, pblk, 0, stream>>>(Vsrc, dinv, deg, BT, mode);
    k_gemm<<<NN / 32, 128, 0, stream>>>(adj_bf, BT, Vsrc, dinv, Out, mode);
  };

  // GCN chain (pre-activation propagated) and scattering chain, interleaved
  pass(X,   hA,  0);
  pass(X,   px,  1);
  pass(hA,  hA2, 0);
  pass(px,  p2x, 1);
  pass(hA2, hA3, 0);
  pass(p2x, p3x, 1);
  pass(p3x, p4x, 1);

  k_fuse<<<NN, 128, 0, stream>>>(X, hA, hA2, hA3, px, p2x, p4x, a, hp);
  k_mlp<<<NN, 128, 0, stream>>>(hp, W1, b1, tmp);
  k_mlp<<<NN, 128, 0, stream>>>(tmp, W2, b2, (float*)d_out);
}
